// SCTConv_13503377179091
// MI455X (gfx1250) — compile-verified, matrix operand order checked by reading, not run on hardware
//
#include <hip/hip_runtime.h>
#include <hip/hip_bf16.h>
#include <math.h>

#define NV 4096
#define FV 64
#define BV 8
#define LEAKY_SLOPE 0.01f

typedef __attribute__((ext_vector_type(16))) __bf16 v16bf;
typedef __attribute__((ext_vector_type(8)))  __bf16 v8bf;
typedef __attribute__((ext_vector_type(8)))  float  v8f;

__device__ __forceinline__ float leakyf(float x) { return x > 0.f ? x : LEAKY_SLOPE * x; }

__device__ __forceinline__ v8f zero8() {
  v8f z;
#pragma unroll
  for (int i = 0; i < 8; ++i) z[i] = 0.f;
  return z;
}

// ---------------------------------------------------------------------------
// Kernel 1: row sums of adj -> dinv_sqrt (GCN, rowsum+1) and dinv (scattering)
// one wave per row, 8 rows per 256-thread block
// ---------------------------------------------------------------------------
__global__ __launch_bounds__(256) void rowsum_kernel(const float* __restrict__ adj,
                                                     float* __restrict__ dss,
                                                     float* __restrict__ dvv) {
  const int wave = threadIdx.x >> 5, lane = threadIdx.x & 31;
  const size_t row = (size_t)blockIdx.x * 8 + wave;   // over B*N
  const float* p = adj + row * NV;
  float s = 0.f;
  for (int j = lane * 4; j < NV; j += 32 * 4) {
    float4 v = *(const float4*)(p + j);
    s += v.x + v.y + v.z + v.w;
  }
#pragma unroll
  for (int m = 16; m >= 1; m >>= 1) s += __shfl_xor(s, m);
  if (lane == 0) {
    dss[row] = rsqrtf(s + 1.0f);   // rowsum(adj + I) = rowsum(adj) + 1
    dvv[row] = 1.0f / s;
  }
}

// ---------------------------------------------------------------------------
// Kernel 2: build initial transposed bf16 operands
//   HtG[b][f][i] = bf16(dinv_sqrt[b,i] * X[b,i,f])
//   HtP[b][f][i] = bf16(dinv[b,i]      * X[b,i,f])
// ---------------------------------------------------------------------------
__global__ __launch_bounds__(256) void prep_kernel(const float* __restrict__ X,
                                                   const float* __restrict__ dss,
                                                   const float* __restrict__ dvv,
                                                   __bf16* __restrict__ HtG,
                                                   __bf16* __restrict__ HtP) {
  const size_t t = (size_t)blockIdx.x * blockDim.x + threadIdx.x;  // over B*N*F
  const int f = (int)(t & 63);
  const int i = (int)((t >> 6) & (NV - 1));
  const int b = (int)(t >> 18);
  const float x = X[t];
  const size_t o = ((size_t)b * FV + f) * NV + i;
  HtG[o] = (__bf16)(dss[(size_t)b * NV + i] * x);
  HtP[o] = (__bf16)(dvv[(size_t)b * NV + i] * x);
}

// ---------------------------------------------------------------------------
// Kernel 3: fused diffusion step (DUAL templated: no branch in the k-loop).
//   Scattering: p_new = 0.5*p_prev + 0.5 * adj @ (dinv*p_prev)
//   GCN (dual): h_new = (adj @ (ds*h_prev) + ds*h_prev) * ds      (A = adj + I)
// B operands consumed as transposed bf16 [B,64,N]; all B fragments are loaded
// before the WMMA chain so the waits pipeline (loadcnt 6/4/2/0 staggering).
// Epilogue emits the next step's transposed bf16 operands with 16B stores.
// ---------------------------------------------------------------------------
template <int DUAL>
__global__ __launch_bounds__(256) void step_kernel(
    const float* __restrict__ adj,
    const __bf16* __restrict__ HtGin, const __bf16* __restrict__ HtPin,
    const float* __restrict__ h_prev, const float* __restrict__ p_prev,
    const float* __restrict__ dss, const float* __restrict__ dvv,
    float* __restrict__ h_out, __bf16* __restrict__ HtGout,
    float* __restrict__ p_out, __bf16* __restrict__ HtPout) {
  const int wave = threadIdx.x >> 5, lane = threadIdx.x & 31;
  const int lrow = lane & 15, lhalf = lane >> 4;
  const int b = blockIdx.y;
  const int rowBase = blockIdx.x * 128 + wave * 16;

  v8f accP[4], accG[DUAL ? 4 : 1];
#pragma unroll
  for (int i = 0; i < 4; ++i) accP[i] = zero8();
  if (DUAL) {
#pragma unroll
    for (int i = 0; i < 4; ++i) accG[i] = zero8();
  }

  const float*  aRow = adj + ((size_t)b * NV + rowBase + lrow) * NV;
  const __bf16* hpB  = HtPin + ((size_t)b * FV + lrow) * NV;
  const __bf16* hgB  = DUAL ? (HtGin + ((size_t)b * FV + lrow) * NV) : nullptr;

  for (int k0 = 0; k0 < NV; k0 += 32) {
    const int kb = k0 + lhalf * 8;
    // Stream-prefetch adj one k-tile ahead (speculative; OOB is dropped).
    __builtin_prefetch(aRow + kb + 32, 0, 0);

    // ---- issue ALL loads up front (one clause, staggered waits later) ----
    float4 r0 = *(const float4*)(aRow + kb);
    float4 r1 = *(const float4*)(aRow + kb + 4);
    float4 r2 = *(const float4*)(aRow + kb + 16);
    float4 r3 = *(const float4*)(aRow + kb + 20);

    v8bf pLo[4], pHi[4];
#pragma unroll
    for (int nt = 0; nt < 4; ++nt) {
      const __bf16* hr = hpB + (size_t)nt * 16 * NV + kb;
      pLo[nt] = *(const v8bf*)(hr);
      pHi[nt] = *(const v8bf*)(hr + 16);
    }
    v8bf gLo[DUAL ? 4 : 1], gHi[DUAL ? 4 : 1];
    if (DUAL) {
#pragma unroll
      for (int nt = 0; nt < 4; ++nt) {
        const __bf16* hr = hgB + (size_t)nt * 16 * NV + kb;
        gLo[nt] = *(const v8bf*)(hr);
        gHi[nt] = *(const v8bf*)(hr + 16);
      }
    }

    // ---- A fragment: f32 -> packed bf16 (two runs of 8 contiguous K) ----
    v16bf af;
    af[0]  = (__bf16)r0.x; af[1]  = (__bf16)r0.y; af[2]  = (__bf16)r0.z; af[3]  = (__bf16)r0.w;
    af[4]  = (__bf16)r1.x; af[5]  = (__bf16)r1.y; af[6]  = (__bf16)r1.z; af[7]  = (__bf16)r1.w;
    af[8]  = (__bf16)r2.x; af[9]  = (__bf16)r2.y; af[10] = (__bf16)r2.z; af[11] = (__bf16)r2.w;
    af[12] = (__bf16)r3.x; af[13] = (__bf16)r3.y; af[14] = (__bf16)r3.z; af[15] = (__bf16)r3.w;

    // ---- WMMA chain ----
#pragma unroll
    for (int nt = 0; nt < 4; ++nt) {
      v16bf bfv = __builtin_shufflevector(pLo[nt], pHi[nt],
                                          0, 1, 2, 3, 4, 5, 6, 7,
                                          8, 9, 10, 11, 12, 13, 14, 15);
      accP[nt] = __builtin_amdgcn_wmma_f32_16x16x32_bf16(
          false, af, false, bfv, (short)0, accP[nt], false, false);
    }
    if (DUAL) {
#pragma unroll
      for (int nt = 0; nt < 4; ++nt) {
        v16bf bfv = __builtin_shufflevector(gLo[nt], gHi[nt],
                                            0, 1, 2, 3, 4, 5, 6, 7,
                                            8, 9, 10, 11, 12, 13, 14, 15);
        accG[nt] = __builtin_amdgcn_wmma_f32_16x16x32_bf16(
            false, af, false, bfv, (short)0, accG[nt], false, false);
      }
    }
  }

  // Epilogue. C layout: lane owns 8 consecutive rows r0i..r0i+7 at column f.
  const int r0i = rowBase + lhalf * 8;
  float4 d0 = *(const float4*)(dvv + (size_t)b * NV + r0i);
  float4 d1 = *(const float4*)(dvv + (size_t)b * NV + r0i + 4);
  float dvr[8] = {d0.x, d0.y, d0.z, d0.w, d1.x, d1.y, d1.z, d1.w};

#pragma unroll
  for (int nt = 0; nt < 4; ++nt) {
    const int f = nt * 16 + lrow;
    v8bf pk;
#pragma unroll
    for (int r = 0; r < 8; ++r) {
      const size_t idx = ((size_t)b * NV + r0i + r) * FV + f;
      const float pn = 0.5f * p_prev[idx] + 0.5f * accP[nt][r];
      p_out[idx] = pn;
      pk[r] = (__bf16)(dvr[r] * pn);
    }
    if (HtPout) *(v8bf*)(HtPout + ((size_t)b * FV + f) * NV + r0i) = pk;
  }

  if (DUAL) {
    float4 s0 = *(const float4*)(dss + (size_t)b * NV + r0i);
    float4 s1 = *(const float4*)(dss + (size_t)b * NV + r0i + 4);
    float dsr[8] = {s0.x, s0.y, s0.z, s0.w, s1.x, s1.y, s1.z, s1.w};
#pragma unroll
    for (int nt = 0; nt < 4; ++nt) {
      const int f = nt * 16 + lrow;
      v8bf pk;
#pragma unroll
      for (int r = 0; r < 8; ++r) {
        const size_t idx = ((size_t)b * NV + r0i + r) * FV + f;
        const float hn = (accG[nt][r] + dsr[r] * h_prev[idx]) * dsr[r];
        h_out[idx] = hn;
        pk[r] = (__bf16)(dsr[r] * hn);
      }
      if (HtGout) *(v8bf*)(HtGout + ((size_t)b * FV + f) * NV + r0i) = pk;
    }
  }
}

// ---------------------------------------------------------------------------
// Kernel 4: attention over 6 branches, one wave per (b,n) row. Lane l handles
// features l and l+32; wave32 xor all-reduce for the dot products.
// ---------------------------------------------------------------------------
__global__ __launch_bounds__(256) void attn_kernel(
    const float* __restrict__ X, const float* __restrict__ h1,
    const float* __restrict__ h2, const float* __restrict__ p1,
    const float* __restrict__ p2, const float* __restrict__ p4,
    const float* __restrict__ p8, const float* __restrict__ avec,
    const int* __restrict__ momentp, float* __restrict__ hprime) {
  const int wave = threadIdx.x >> 5, lane = threadIdx.x & 31;
  const size_t row = (size_t)blockIdx.x * 8 + wave;   // over B*N
  const size_t base = row * FV;
  const float mom = (float)(*momentp);

  float v[6][2];
  float eSh = 0.f, ep[6];
#pragma unroll
  for (int k = 0; k < 6; ++k) ep[k] = 0.f;

#pragma unroll
  for (int s = 0; s < 2; ++s) {
    const int f = lane + 32 * s;
    const float x  = X[base + f];
    const float a0 = avec[f], a1 = avec[FV + f];
    eSh += fmaxf(x, 0.f) * a0;
    const float q1 = p1[base + f], q2 = p2[base + f];
    const float q4 = p4[base + f], q8 = p8[base + f];
    float t;
    t = leakyf(h1[base + f]);                                      v[0][s] = t;
    t = leakyf(h2[base + f]);                                      v[1][s] = t;
    t = (mom == 1.f) ? fabsf(x - q1)  : powf(fabsf(x - q1),  mom); v[2][s] = t;
    t = (mom == 1.f) ? fabsf(q1 - q2) : powf(fabsf(q1 - q2), mom); v[3][s] = t;
    t = (mom == 1.f) ? fabsf(q2 - q4) : powf(fabsf(q2 - q4), mom); v[4][s] = t;
    t = (mom == 1.f) ? fabsf(q4 - q8) : powf(fabsf(q4 - q8), mom); v[5][s] = t;
#pragma unroll
    for (int k = 0; k < 6; ++k) ep[k] += fmaxf(v[k][s], 0.f) * a1;
  }

#pragma unroll
  for (int m = 16; m >= 1; m >>= 1) {
    eSh += __shfl_xor(eSh, m);
#pragma unroll
    for (int k = 0; k < 6; ++k) ep[k] += __shfl_xor(ep[k], m);
  }

  float e[6], mx = -3.4e38f;
#pragma unroll
  for (int k = 0; k < 6; ++k) { e[k] = eSh + ep[k]; mx = fmaxf(mx, e[k]); }
  float sum = 0.f, w[6];
#pragma unroll
  for (int k = 0; k < 6; ++k) { w[k] = __expf(e[k] - mx); sum += w[k]; }
  const float inv = 1.f / (6.f * sum);   // softmax * mean over 6 branches

#pragma unroll
  for (int s = 0; s < 2; ++s) {
    float acc = 0.f;
#pragma unroll
    for (int k = 0; k < 6; ++k) acc += w[k] * v[k][s];
    hprime[base + lane + 32 * s] = acc * inv;
  }
}

// ---------------------------------------------------------------------------
// Kernel 5: out = leaky(leaky(hprime@W1^T + b1)@W2^T + b2). 16 rows per block,
// W1/W2 in LDS with pad-65 to avoid bank conflicts.
// ---------------------------------------------------------------------------
__global__ __launch_bounds__(256) void mlp_kernel(
    const float* __restrict__ hp, const float* __restrict__ W1,
    const float* __restrict__ b1, const float* __restrict__ W2,
    const float* __restrict__ b2, float* __restrict__ out) {
  __shared__ float sW1[64 * 65];
  __shared__ float sW2[64 * 65];
  __shared__ float sh[16 * 64];
  __shared__ float st[16 * 64];

  for (int i = threadIdx.x; i < 4096; i += 256) {
    sW1[(i >> 6) * 65 + (i & 63)] = W1[i];
    sW2[(i >> 6) * 65 + (i & 63)] = W2[i];
  }
  const size_t rb = (size_t)blockIdx.x * 16;
  for (int i = threadIdx.x; i < 16 * 64; i += 256) sh[i] = hp[rb * FV + i];
  __syncthreads();

  const int f = threadIdx.x & 63, rq = threadIdx.x >> 6;
  const float bb1 = b1[f];
#pragma unroll
  for (int j = 0; j < 4; ++j) {
    const int r = rq * 4 + j;
    float s = bb1;
#pragma unroll 8
    for (int c = 0; c < 64; ++c) s += sh[r * 64 + c] * sW1[f * 65 + c];
    st[r * 64 + f] = leakyf(s);
  }
  __syncthreads();
  const float bb2 = b2[f];
#pragma unroll
  for (int j = 0; j < 4; ++j) {
    const int r = rq * 4 + j;
    float s = bb2;
#pragma unroll 8
    for (int c = 0; c < 64; ++c) s += st[r * 64 + c] * sW2[f * 65 + c];
    out[(rb + r) * FV + f] = leakyf(s);
  }
}

// ---------------------------------------------------------------------------
extern "C" void kernel_launch(void* const* d_in, const int* in_sizes, int n_in,
                              void* d_out, int out_size, void* d_ws, size_t ws_size,
                              hipStream_t stream) {
  const float* X    = (const float*)d_in[0];
  const float* adj  = (const float*)d_in[1];
  const float* W1   = (const float*)d_in[2];
  const float* b1   = (const float*)d_in[3];
  const float* W2   = (const float*)d_in[4];
  const float* b2   = (const float*)d_in[5];
  const float* avec = (const float*)d_in[6];
  const int*   mom  = (const int*)d_in[7];

  char* ws = (char*)d_ws;
  size_t off = 0;
  auto carve = [&](size_t bytes) -> char* {
    char* p = ws + off;
    off += (bytes + 255) & ~(size_t)255;
    return p;
  };
  const size_t vecB  = (size_t)BV * NV * sizeof(float);              // 128 KB
  const size_t featB = (size_t)BV * NV * FV * sizeof(float);         // 8 MB
  const size_t htB   = (size_t)BV * FV * NV * sizeof(__bf16);        // 4 MB

  float* dss = (float*)carve(vecB);
  float* dvv = (float*)carve(vecB);
  float* h1  = (float*)carve(featB);
  float* h2  = (float*)carve(featB);
  float* p1  = (float*)carve(featB);
  float* p2  = (float*)carve(featB);
  float* p4  = (float*)carve(featB);
  float* pA  = (float*)carve(featB);
  float* pB  = (float*)carve(featB);
  float* hpr = (float*)carve(featB);
  __bf16* HtG0 = (__bf16*)carve(htB);
  __bf16* HtG1 = (__bf16*)carve(htB);
  __bf16* HtP0 = (__bf16*)carve(htB);
  __bf16* HtP1 = (__bf16*)carve(htB);

  rowsum_kernel<<<BV * NV / 8, 256, 0, stream>>>(adj, dss, dvv);
  prep_kernel<<<(BV * NV * FV) / 256, 256, 0, stream>>>(X, dss, dvv, HtG0, HtP0);

  dim3 g(NV / 128, BV);
  // step 1 (GCN + scattering fused)
  step_kernel<1><<<g, 256, 0, stream>>>(adj, HtG0, HtP0, X, X, dss, dvv,
                                        h1, HtG1, p1, HtP1);
  // step 2 (GCN + scattering fused; hA3 is unused by the reference -> stop GCN)
  step_kernel<1><<<g, 256, 0, stream>>>(adj, HtG1, HtP1, h1, p1, dss, dvv,
                                        h2, nullptr, p2, HtP0);
  // steps 3..8: scattering only
  step_kernel<0><<<g, 256, 0, stream>>>(adj, nullptr, HtP0, nullptr, p2, dss, dvv,
                                        nullptr, nullptr, pA, HtP1);  // P^3
  step_kernel<0><<<g, 256, 0, stream>>>(adj, nullptr, HtP1, nullptr, pA, dss, dvv,
                                        nullptr, nullptr, p4, HtP0);  // P^4
  step_kernel<0><<<g, 256, 0, stream>>>(adj, nullptr, HtP0, nullptr, p4, dss, dvv,
                                        nullptr, nullptr, pB, HtP1);  // P^5
  step_kernel<0><<<g, 256, 0, stream>>>(adj, nullptr, HtP1, nullptr, pB, dss, dvv,
                                        nullptr, nullptr, pA, HtP0);  // P^6
  step_kernel<0><<<g, 256, 0, stream>>>(adj, nullptr, HtP0, nullptr, pA, dss, dvv,
                                        nullptr, nullptr, pB, HtP1);  // P^7
  step_kernel<0><<<g, 256, 0, stream>>>(adj, nullptr, HtP1, nullptr, pB, dss, dvv,
                                        nullptr, nullptr, pA, nullptr);  // P^8

  attn_kernel<<<BV * NV / 8, 256, 0, stream>>>(X, h1, h2, p1, p2, p4, pA,
                                               avec, mom, hpr);
  mlp_kernel<<<BV * NV / 16, 256, 0, stream>>>(hpr, W1, b1, W2, b2,
                                               (float*)d_out);
}
